// BayesianSkipgram_66005057405537
// MI455X (gfx1250) — matrix operand order repaired and checked by reading, not compiled, not run on hardware
//
#include <hip/hip_runtime.h>
#include <math.h>

// ---------------------------------------------------------------------------
// BayesianSkipgram forward on MI455X (gfx1250, wave32, WMMA).
//   V=100000, E=256, C=8192, NEG=10.  Output: single float.
// Pipeline (all on `stream`, in order):
//   k_zero       : zero h2 accumulators
//   k_cvt_mw     : M_w f32 -> bf16 copy (B operand for WMMA, 128KB, L2-hot)
//   k_rw         : Rw = M_w @ emb[x] + M_b ; h1 = C * relu(Rw)
//   k_gemm_rowsum: WMMA bf16 GEMM of gathered emb rows x M_w^T, fused
//                  bias+relu+column-sum -> h2 (Rc never hits HBM)
//   k_musigma    : mu, sigma=softplus(W h), inv2v, base, kl_main -> out[0]
//   k_negkl      : neg_kl[10]
//   k_poskl      : pos_kl per context row + hinge vs neg_kl, atomicAdd out[0]
// ---------------------------------------------------------------------------

#define E_DIM 256
#define TWO_E 512

typedef __attribute__((ext_vector_type(16))) __bf16 v16bf;
typedef __attribute__((ext_vector_type(8)))  __bf16 v8bf;
typedef __attribute__((ext_vector_type(8)))  float  v8f;

// ---------------- small helpers ----------------
__device__ __forceinline__ float wave_reduce_sum(float v) {
  #pragma unroll
  for (int off = 16; off > 0; off >>= 1) v += __shfl_down(v, off, 32);
  return v;
}

__device__ __forceinline__ float softplus_f(float x) {
  return (x > 20.0f) ? x : log1pf(expf(x));
}

// ---------------- kernel: zero h2 ----------------
__global__ void k_zero(float* __restrict__ h2) {
  h2[threadIdx.x] = 0.0f;
}

// ---------------- kernel: convert M_w to bf16 ----------------
__global__ void k_cvt_mw(const float* __restrict__ mw, __bf16* __restrict__ mwbf) {
  int i = blockIdx.x * 256 + threadIdx.x;
  mwbf[i] = (__bf16)mw[i];
}

// ---------------- kernel: Rw and h1 = C*relu(Rw) ----------------
__global__ void k_rw(const int* __restrict__ x, const float* __restrict__ emb,
                     const float* __restrict__ mw, const float* __restrict__ mb,
                     float* __restrict__ h1, float ctx_count) {
  __shared__ float ex[E_DIM];
  const int tid = threadIdx.x;
  const int xi = x[0];
  ex[tid] = emb[(size_t)xi * E_DIM + tid];
  __syncthreads();
  float s = mb[tid];
  const float* row = mw + (size_t)tid * E_DIM;
  #pragma unroll 4
  for (int k = 0; k < E_DIM; ++k) s += row[k] * ex[k];
  h1[tid] = ctx_count * fmaxf(s, 0.0f);
}

// ---------------- kernel: WMMA GEMM + fused relu column-sum ----------------
// One block = 16 context rows x all 256 output cols. 16 waves; wave w owns
// the 16-col tile [16w, 16w+16). K loop: 8 x v_wmma_f32_16x16x32_bf16.
__global__ __launch_bounds__(512)
void k_gemm_rowsum(const float* __restrict__ emb, const int* __restrict__ context,
                   const __bf16* __restrict__ mwbf, const float* __restrict__ mb,
                   float* __restrict__ h2) {
  __shared__ __bf16 As[16][E_DIM + 8];   // +8 bf16 pad: rows land on distinct banks
  __shared__ int rows[16];

  const int tid = threadIdx.x;
  if (tid < 16) rows[tid] = context[blockIdx.x * 16 + tid];
  __syncthreads();

  // gather 16 embedding rows, convert f32 -> bf16 into LDS
  #pragma unroll
  for (int i = tid; i < 16 * E_DIM; i += 512) {
    const int r = i >> 8, k = i & (E_DIM - 1);
    As[r][k] = (__bf16)emb[(size_t)rows[r] * E_DIM + k];
  }
  __syncthreads();

  const int wave = tid >> 5;
  const int lane = tid & 31;
  const int hh   = lane >> 4;          // half-wave select
  const int m    = lane & 15;          // A row for this lane
  const int n    = (wave << 4) | m;    // global output column (B/D col)

  v8f acc = {};
  #pragma unroll
  for (int kb = 0; kb < E_DIM; kb += 32) {
    // A 16x32 bf16 fragment: K runs [kb+16h, +16)
    v8bf a0 = *(const v8bf*)&As[m][kb + 16 * hh];
    v8bf a1 = *(const v8bf*)&As[m][kb + 16 * hh + 8];
    v16bf a = __builtin_shufflevector(a0, a1, 0,1,2,3,4,5,6,7,8,9,10,11,12,13,14,15);
    // B 32x16 bf16 fragment: B[k][n] = M_w[n][k]; lane holds K = kb+16h+e
    v16bf b = *(const v16bf*)&mwbf[(size_t)n * E_DIM + kb + 16 * hh];
    acc = __builtin_amdgcn_wmma_f32_16x16x32_bf16(
        /*neg_a=*/false, a, /*neg_b=*/false, b,
        /*c_mod=*/(short)0, acc, /*reuse_a=*/false, /*reuse_b=*/false);
  }

  // D element r -> row (r + 8h), col n. Fuse bias + relu + 16-row column sum.
  const float bias = mb[n];
  float cs = 0.0f;
  #pragma unroll
  for (int r = 0; r < 8; ++r) cs += fmaxf(acc[r] + bias, 0.0f);
  cs += __shfl_xor(cs, 16, 32);        // combine the two row-halves
  if (hh == 0) atomicAdd(&h2[n], cs);  // one atomic per (block, column)
}

// ---------------- kernel: mu, sigma, inv2v, base, kl_main ----------------
__global__ void k_musigma(const float* __restrict__ hvec,   // [512] = {h1,h2}
                          const float* __restrict__ uw, const float* __restrict__ ub,
                          const float* __restrict__ ww, const float* __restrict__ wb,
                          const int* __restrict__ x,
                          const float* __restrict__ prior_mus,
                          const float* __restrict__ prior_sigmas,
                          float* __restrict__ mu, float* __restrict__ inv2v,
                          float* __restrict__ base_out, float* __restrict__ out) {
  __shared__ float hs[TWO_E];
  __shared__ float ra[E_DIM], rb[E_DIM];
  const int tid = threadIdx.x;
  hs[tid]         = hvec[tid];
  hs[tid + E_DIM] = hvec[tid + E_DIM];
  __syncthreads();

  float mu_e = ub[tid], sg = wb[tid];
  const float* urow = uw + (size_t)tid * TWO_E;
  const float* wrow = ww + (size_t)tid * TWO_E;
  #pragma unroll 4
  for (int j = 0; j < TWO_E; ++j) {
    const float hj = hs[j];
    mu_e += urow[j] * hj;
    sg   += wrow[j] * hj;
  }
  const float sigma = softplus_f(sg);
  const float lgs   = logf(sigma);
  mu[tid]    = mu_e;
  inv2v[tid] = 0.5f / (sigma * sigma);

  const int xi   = x[0];
  const float pm = prior_mus[(size_t)xi * E_DIM + tid];
  const float ps = prior_sigmas[(size_t)xi * E_DIM + tid];
  const float vp = ps * ps;
  const float dm = pm - mu_e;
  ra[tid] = logf(vp) - lgs + sigma / vp + dm * dm / vp;  // kl_main terms
  rb[tid] = 0.5f - lgs;                                  // _kl_rows base terms
  __syncthreads();
  #pragma unroll
  for (int s = 128; s > 0; s >>= 1) {
    if (tid < s) { ra[tid] += ra[tid + s]; rb[tid] += rb[tid + s]; }
    __syncthreads();
  }
  if (tid == 0) {
    out[0]      = 0.5f * (ra[0] - (float)E_DIM);  // kl_main (likelihood added later)
    base_out[0] = rb[0];
  }
}

// ---------------- kernel: neg_kl rows ----------------
__global__ void k_negkl(const int* __restrict__ negs,
                        const float* __restrict__ prior_mus,
                        const float* __restrict__ prior_sigmas,
                        const float* __restrict__ mu, const float* __restrict__ inv2v,
                        const float* __restrict__ base, float* __restrict__ negkl) {
  const int wave = threadIdx.x >> 5;
  const int lane = threadIdx.x & 31;
  const int idx  = negs[wave];
  float v = 0.0f;
  #pragma unroll
  for (int e = lane; e < E_DIM; e += 32) {
    const float m = fabsf(prior_mus[(size_t)idx * E_DIM + e]);
    const float s = fabsf(prior_sigmas[(size_t)idx * E_DIM + e]);
    const float d = mu[e] - m;
    v += inv2v[e] * (d * d - s * s) + logf(s);
  }
  v = wave_reduce_sum(v);
  if (lane == 0) negkl[wave] = v + base[0];
}

// ---------------- kernel: pos_kl per context row + hinge sum ----------------
__global__ __launch_bounds__(256)
void k_poskl(const int* __restrict__ context,
             const float* __restrict__ prior_mus, const float* __restrict__ prior_sigmas,
             const float* __restrict__ mu, const float* __restrict__ inv2v,
             const float* __restrict__ base, const float* __restrict__ negkl,
             int n_neg, float* __restrict__ out) {
  const int wave = threadIdx.x >> 5;
  const int lane = threadIdx.x & 31;
  const int row  = blockIdx.x * 8 + wave;
  const int idx  = context[row];
  float v = 0.0f;
  #pragma unroll
  for (int e = lane; e < E_DIM; e += 32) {
    const float m = fabsf(prior_mus[(size_t)idx * E_DIM + e]);
    const float s = fabsf(prior_sigmas[(size_t)idx * E_DIM + e]);
    const float d = mu[e] - m;
    v += inv2v[e] * (d * d - s * s) + logf(s);
  }
  v = wave_reduce_sum(v);
  if (lane == 0) {
    const float p = v + base[0];
    float acc = 0.0f;
    for (int j = 0; j < n_neg; ++j)
      acc += fmaxf(0.0f, p - negkl[j] + 1.0f);
    atomicAdd(out, acc);
  }
}

// ---------------------------------------------------------------------------
extern "C" void kernel_launch(void* const* d_in, const int* in_sizes, int n_in,
                              void* d_out, int out_size, void* d_ws, size_t ws_size,
                              hipStream_t stream) {
  const int*   x       = (const int*)  d_in[0];
  const int*   context = (const int*)  d_in[1];
  const int*   negs    = (const int*)  d_in[2];
  const float* emb     = (const float*)d_in[3];
  const float* M_w     = (const float*)d_in[4];
  const float* M_b     = (const float*)d_in[5];
  const float* U_w     = (const float*)d_in[6];
  const float* U_b     = (const float*)d_in[7];
  const float* W_w     = (const float*)d_in[8];
  const float* W_b     = (const float*)d_in[9];
  const float* p_mus   = (const float*)d_in[10];
  const float* p_sig   = (const float*)d_in[11];
  float* out = (float*)d_out;

  const int C   = in_sizes[1];   // 8192
  const int NEG = in_sizes[2];   // 10

  // workspace layout (floats)
  float* wsf   = (float*)d_ws;
  float* h1    = wsf;            // [256]  (h1,h2 contiguous = h[512])
  float* h2    = wsf + 256;      // [256]
  float* mu    = wsf + 512;      // [256]
  float* inv2v = wsf + 768;      // [256]
  float* base  = wsf + 1024;     // [1]
  float* negkl = wsf + 1032;     // [NEG]
  __bf16* mwbf = (__bf16*)(wsf + 2048);  // [256*256] bf16 (128 KB)

  k_zero<<<1, 256, 0, stream>>>(h2);
  k_cvt_mw<<<E_DIM, 256, 0, stream>>>(M_w, mwbf);
  k_rw<<<1, 256, 0, stream>>>(x, emb, M_w, M_b, h1, (float)C);
  k_gemm_rowsum<<<C / 16, 512, 0, stream>>>(emb, context, mwbf, M_b, h2);
  k_musigma<<<1, 256, 0, stream>>>(h1, U_w, U_b, W_w, W_b, x, p_mus, p_sig,
                                   mu, inv2v, base, out);
  k_negkl<<<1, 32 * NEG, 0, stream>>>(negs, p_mus, p_sig, mu, inv2v, base, negkl);
  k_poskl<<<C / 8, 256, 0, stream>>>(context, p_mus, p_sig, mu, inv2v, base,
                                     negkl, NEG, out);
}
